// HMPNN_29661044146419
// MI455X (gfx1250) — matrix operand order repaired and checked by reading, not compiled, not run on hardware
//
#include <hip/hip_runtime.h>
#include <cstdint>

// ---------------------------------------------------------------------------
// HMPNN forward for MI455X (gfx1250, wave32, WMMA).
//  - Never materialize per-edge [96x96] NNConv weights (saves ~755MB HBM
//    traffic); edge-weight generation fused into WMMA A-fragments.
//  - All B operands pre-packed into WMMA fragment-major layout so each lane's
//    fragment is one contiguous 32B load; block-shared B strips staged in LDS
//    (async global->LDS when the toolchain exposes the gfx1250 builtin).
// ---------------------------------------------------------------------------

typedef __attribute__((ext_vector_type(16))) _Float16 v16h;
typedef __attribute__((ext_vector_type(8)))  float    v8f;
typedef __attribute__((ext_vector_type(4)))  int      v4i_;

#define NA   8192
#define NB   16384
#define NF   2048
#define NEF  4096
#define NG   256
#define HID  96
#define OUTD 12
#define AIN  40
#define BIN  10
#define FNIN 82
#define FEIN 130
#define MIN_ 167
#define REP  359   // 2*HID + MIN_
#define REPP 384   // REP padded to mult of 32

#if defined(__has_builtin)
#if __has_builtin(__builtin_amdgcn_global_load_async_to_lds_b128) && \
    __has_builtin(__builtin_amdgcn_s_wait_asynccnt)
#define ASYNC_LDS 1
#endif
#endif

// ---------------------------- small utility kernels ------------------------

__global__ void k_fill_f32(float* __restrict__ p, float v, int n) {
    int i = blockIdx.x * blockDim.x + threadIdx.x;
    if (i < n) p[i] = v;
}
__global__ void k_fill_f16(_Float16* __restrict__ p, int n) {
    int i = blockIdx.x * blockDim.x + threadIdx.x;
    if (i < n) p[i] = (_Float16)0.0f;
}
// f32 -> f16 with layout control (K-padding / concatenation), row-major dst
__global__ void k_cvt_f16(const float* __restrict__ src, _Float16* __restrict__ dst,
                          int M, int K, int ld_src, int ld_dst, int col_off) {
    int i = blockIdx.x * blockDim.x + threadIdx.x;
    if (i >= M * K) return;
    int r = i / K, c = i - r * K;
    dst[(size_t)r * ld_dst + col_off + c] = (_Float16)src[(size_t)r * ld_src + c];
}

// Pack a (K x N) row-major f32 weight into WMMA fragment-major f16:
//   dst[(((nt*KC + kc)*32 + lane)*16) + j] = W[kc*32 + (lane>>4)*16 + j, nt*16 + (lane&15)]
// so each lane's 16-half B fragment is contiguous (one 32B load / 2x b128).
__global__ void k_pack_b(const float* __restrict__ src, _Float16* __restrict__ dst,
                         int K, int N, int Kpad, int Npad) {
    int idx = blockIdx.x * blockDim.x + threadIdx.x;
    if (idx >= Kpad * Npad) return;
    int j    = idx & 15;
    int lane = (idx >> 4) & 31;
    int rest = idx >> 9;
    int KC   = Kpad >> 5;
    int kc   = rest % KC;
    int nt   = rest / KC;
    int k = kc * 32 + ((lane >> 4) << 4) + j;
    int n = nt * 16 + (lane & 15);
    float v = (k < K && n < N) ? src[(size_t)k * N + n] : 0.f;
    dst[idx] = (_Float16)v;
}
// Same packing for NNConv: virtual B row k=(kk*96+i) is edge_W[kk][i*96+o]
// for kk<Kf and edge_b[i*96+o] for the bias block kk==Kf. K=(Kf+1)*96, N=96.
__global__ void k_pack_edgeW(const float* __restrict__ W, const float* __restrict__ b,
                             _Float16* __restrict__ dst, int Kf) {
    int K = (Kf + 1) * HID;
    int idx = blockIdx.x * blockDim.x + threadIdx.x;
    if (idx >= K * HID) return;
    int j    = idx & 15;
    int lane = (idx >> 4) & 31;
    int rest = idx >> 9;
    int KC   = K >> 5;
    int kc   = rest % KC;
    int nt   = rest / KC;
    int k = kc * 32 + ((lane >> 4) << 4) + j;
    int n = nt * 16 + (lane & 15);
    int kk = k / HID, i = k - kk * HID;
    float v = (kk < Kf) ? W[(size_t)kk * HID * HID + i * HID + n] : b[i * HID + n];
    dst[idx] = (_Float16)v;
}

// ------------------------------ WMMA fragments ------------------------------

__device__ inline v16h load_a_frag(const _Float16* __restrict__ A, int lda, int k0, int lane) {
    // 16x32 f16 A tile (ISA 7.12.2): lane-half selects K sub-block.
    int r  = lane & 15;
    int kb = (lane >> 4) << 4;
    const _Float16* p = A + (size_t)r * lda + k0 + kb;
    v16h a;
#pragma unroll
    for (int j = 0; j < 8; ++j) { a[j] = p[j]; a[8 + j] = p[8 + j]; }
    return a;
}

// Stage one 16-half (32B) chunk into LDS; async global->LDS when available.
__device__ inline void stage_chunk(const _Float16* __restrict__ g, _Float16* l) {
#if ASYNC_LDS
    __builtin_amdgcn_global_load_async_to_lds_b128(
        (v4i_*)(void*)(uintptr_t)g, (v4i_*)(void*)l, 0, 0);
    __builtin_amdgcn_global_load_async_to_lds_b128(
        (v4i_*)(void*)(uintptr_t)(g + 8), (v4i_*)(void*)(l + 8), 0, 0);
#else
    *(v16h*)l = *(const v16h*)g;
#endif
}
__device__ inline void stage_wait() {
#if ASYNC_LDS
    __builtin_amdgcn_s_wait_asynccnt(0);
#endif
    __syncthreads();
}

// ------------------------------- GEMM kernel --------------------------------
// C[M x Nreal] = epilogue(A[M x K]f16 @ Bpacked[K x Npad]f16); K<=384.
// 4 waves/block, wave = one 16-row tile; block's 16-col B strip staged in LDS.
// grid = (Npad/16, M/64), block = 128.
__global__ void k_wmma_gemm(const _Float16* __restrict__ A, int lda,
                            const _Float16* __restrict__ Bp,
                            const float* __restrict__ bias,
                            const float* __restrict__ gamma,
                            const float* __restrict__ beta,
                            float* __restrict__ C, int ldc,
                            int K, int Nreal, int relu) {
    __shared__ __attribute__((aligned(32))) _Float16 smem[12 * 512];  // 12KB max
    const int tid  = threadIdx.x;
    const int lane = tid & 31;
    const int wv   = tid >> 5;
    const int KC   = K >> 5;
    const _Float16* Btile = Bp + (size_t)blockIdx.x * KC * 512;
    for (int t = tid; t < KC * 32; t += 128)
        stage_chunk(Btile + (size_t)t * 16, smem + t * 16);
    stage_wait();

    const int row0 = (blockIdx.y * 4 + wv) * 16;
    const int col0 = blockIdx.x * 16;
    const _Float16* Arow = A + (size_t)row0 * lda;
    v8f acc = {0.f, 0.f, 0.f, 0.f, 0.f, 0.f, 0.f, 0.f};
    for (int kc = 0; kc < KC; ++kc) {
        if (kc + 1 < KC)
            __builtin_prefetch(Arow + (size_t)(lane & 15) * lda + kc * 32 + 32, 0, 1);
        v16h a = load_a_frag(Arow, lda, kc * 32, lane);
        v16h b = *(const v16h*)(smem + ((kc * 32 + lane) << 4));
        acc = __builtin_amdgcn_wmma_f32_16x16x32_f16(false, a, false, b,
                                                     (short)0, acc, false, false);
    }
    int c  = lane & 15;
    int rb = (lane >> 4) << 3;
    int cg = col0 + c;
    if (cg < Nreal) {
        float bi = bias  ? bias[cg]  : 0.f;
        float g  = gamma ? gamma[cg] : 1.f;
        float be = beta  ? beta[cg]  : 0.f;
#pragma unroll
        for (int v = 0; v < 8; ++v) {
            float val = (acc[v] + bi) * g + be;
            if (relu) val = fmaxf(val, 0.f);
            C[(size_t)(row0 + rb + v) * ldc + cg] = val;
        }
    }
}

// ------------------------- fused NNConv message kernel ----------------------
// out[dst[e]] += Z[e] @ W2, Z[e, kk*96+i] = coef(e,kk)*h16[src[e], i],
// coef = ef[e,kk] (kk<Kf) or 1.0 (bias block). A generated in registers.
// USE_LDS=true stages the packed 16-col W2 strip in LDS (atom edges: 33KB);
// fg edges (393KB strip) read packed W2 from global (L2-resident).
template <bool USE_LDS>
__global__ void k_nnconv_msg(const float* __restrict__ ef, int Kf,
                             const _Float16* __restrict__ H,
                             const int* __restrict__ src, const int* __restrict__ dst,
                             const _Float16* __restrict__ Bp,
                             float* __restrict__ out) {
    __shared__ __attribute__((aligned(32))) _Float16 smem[USE_LDS ? 33 * 512 : 32];
    const int tid  = threadIdx.x;
    const int lane = tid & 31;
    const int wv   = tid >> 5;
    const int K    = (Kf + 1) * HID;
    const int KC   = K >> 5;
    const _Float16* Btile = Bp + (size_t)blockIdx.x * KC * 512;
    if (USE_LDS) {
        for (int t = tid; t < KC * 32; t += 128)
            stage_chunk(Btile + (size_t)t * 16, smem + t * 16);
        stage_wait();
    }
    const int e0 = (blockIdx.y * 4 + wv) * 16;
    const int er = e0 + (lane & 15);
    const float*    efp = ef + (size_t)er * Kf;
    const _Float16* hp  = H + (size_t)src[er] * HID;
    const int kb = (lane >> 4) << 4;
    v8f acc = {0.f, 0.f, 0.f, 0.f, 0.f, 0.f, 0.f, 0.f};
    for (int kc = 0; kc < KC; ++kc) {
        int k0 = kc * 32;
        if (!USE_LDS && kc + 1 < KC)
            __builtin_prefetch(Btile + ((size_t)(kc + 1) * 32 + lane) * 16, 0, 1);
        v16h a;
#pragma unroll
        for (int j = 0; j < 8; ++j) {
            int k1 = k0 + kb + j;
            int k2 = k0 + kb + 8 + j;
            int kk1 = k1 / HID, i1 = k1 - kk1 * HID;
            int kk2 = k2 / HID, i2 = k2 - kk2 * HID;
            float c1 = (kk1 < Kf) ? efp[kk1] : 1.0f;
            float c2 = (kk2 < Kf) ? efp[kk2] : 1.0f;
            a[j]     = (_Float16)(c1 * (float)hp[i1]);
            a[8 + j] = (_Float16)(c2 * (float)hp[i2]);
        }
        v16h b = USE_LDS ? *(const v16h*)(smem + ((k0 + lane) << 4))
                         : *(const v16h*)(Btile + ((size_t)(k0 + lane) << 4));
        acc = __builtin_amdgcn_wmma_f32_16x16x32_f16(false, a, false, b,
                                                     (short)0, acc, false, false);
    }
    int c  = lane & 15;
    int rb = (lane >> 4) << 3;
#pragma unroll
    for (int v = 0; v < 8; ++v) {
        int e = e0 + rb + v;
        atomicAdd(&out[(size_t)dst[e] * HID + blockIdx.x * 16 + c], acc[v]);
    }
}

// --------------------------- scatter / segment ops --------------------------

__global__ void k_scatter_add_rows(const float* __restrict__ srcm,
                                   const int* __restrict__ dsti,
                                   float* __restrict__ out, int M) {
    int i = blockIdx.x * blockDim.x + threadIdx.x;
    if (i >= M * HID) return;
    int r = i / HID, c = i - r * HID;
    atomicAdd(&out[(size_t)dsti[r] * HID + c], srcm[i]);
}

__device__ inline void atomicMaxF32(float* addr, float v) {
    if (v >= 0.f) atomicMax((int*)addr, __float_as_int(v));
    else          atomicMin((unsigned int*)addr, (unsigned int)__float_as_int(v));
}

__global__ void k_segment_reduce(const float* __restrict__ h, const int* __restrict__ batch,
                                 int M, float* __restrict__ s, float* __restrict__ m) {
    int i = blockIdx.x * blockDim.x + threadIdx.x;
    if (i >= M * HID) return;
    int r = i / HID, c = i - r * HID;
    int g = batch[r];
    float v = h[i];
    atomicAdd(&s[(size_t)g * HID + c], v);
    atomicMaxF32(&m[(size_t)g * HID + c], v);
}

// rep[g] = [ sum(96) || max(96, -inf->0) || mf(167) ]
__global__ void k_build_rep(const float* __restrict__ s, const float* __restrict__ m,
                            const float* __restrict__ mf, float* __restrict__ rep) {
    int i = blockIdx.x * blockDim.x + threadIdx.x;
    if (i >= NG * REP) return;
    int r = i / REP, c = i - r * REP;
    float v;
    if (c < HID) v = s[(size_t)r * HID + c];
    else if (c < 2 * HID) {
        v = m[(size_t)r * HID + (c - HID)];
        if (v < -3.0e38f) v = 0.f;   // untouched segments (-inf) -> 0
    } else v = mf[(size_t)r * MIN_ + (c - 2 * HID)];
    rep[i] = v;
}

// ------------------------------ GRU elementwise -----------------------------

__device__ inline float sigm(float x) { return 1.f / (1.f + __expf(-x)); }

__global__ void k_gru(const float* __restrict__ gx, const float* __restrict__ gh,
                      float* __restrict__ h, int M) {
    int i = blockIdx.x * blockDim.x + threadIdx.x;
    if (i >= M * HID) return;
    int r = i / HID, c = i - r * HID;
    size_t b3 = (size_t)r * 3 * HID;
    float xr = gx[b3 + c],           hr = gh[b3 + c];
    float xz = gx[b3 + HID + c],     hz = gh[b3 + HID + c];
    float xn = gx[b3 + 2 * HID + c], hn = gh[b3 + 2 * HID + c];
    float rg = sigm(xr + hr);
    float z  = sigm(xz + hz);
    float n  = tanhf(xn + rg * hn);
    h[i] = (1.f - z) * n + z * h[i];
}

// ----------------------------- contextual rescale ---------------------------

__global__ void k_scale_dot(const float* __restrict__ rep, const float* __restrict__ W,
                            const float* __restrict__ b, float* __restrict__ j, int col) {
    int g = blockIdx.x * blockDim.x + threadIdx.x;
    if (g >= NG) return;
    float acc = b[0];
    for (int k = 0; k < REP; ++k) acc += rep[(size_t)g * REP + k] * W[k];
    j[g * 2 + col] = acc;
}
__global__ void k_scale_mlp(const float* __restrict__ j,
                            const float* __restrict__ W1, const float* __restrict__ b1,
                            const float* __restrict__ W2, const float* __restrict__ b2,
                            float* __restrict__ w) {
    int g = blockIdx.x * blockDim.x + threadIdx.x;
    if (g >= NG) return;
    float j0 = j[g * 2], j1 = j[g * 2 + 1];
    float o0 = b2[0], o1 = b2[1];
    for (int h = 0; h < 64; ++h) {
        float hv = fmaxf(j0 * W1[h] + j1 * W1[64 + h] + b1[h], 0.f);
        o0 += hv * W2[h * 2];
        o1 += hv * W2[h * 2 + 1];
    }
    w[g * 2]     = sigm(o0);
    w[g * 2 + 1] = sigm(o1);
}
__global__ void k_scale_apply(float* __restrict__ rep, const float* __restrict__ w, int col) {
    int i = blockIdx.x * blockDim.x + threadIdx.x;
    if (i >= NG * REP) return;
    int r = i / REP;
    rep[i] *= w[r * 2 + col];
}

// --------------------------------- launcher ---------------------------------

extern "C" void kernel_launch(void* const* d_in, const int* in_sizes, int n_in,
                              void* d_out, int out_size, void* d_ws, size_t ws_size,
                              hipStream_t stream) {
    (void)in_sizes; (void)n_in; (void)out_size; (void)ws_size;
    const float* af       = (const float*)d_in[0];
    const float* bf       = (const float*)d_in[1];
    const float* fnf      = (const float*)d_in[2];
    const float* fef      = (const float*)d_in[3];
    const float* mf       = (const float*)d_in[4];
    const int*   a_src    = (const int*)d_in[5];
    const int*   a_dst    = (const int*)d_in[6];
    const int*   f_src    = (const int*)d_in[7];
    const int*   f_dst    = (const int*)d_in[8];
    /* a2f_src = arange(NA) -> implicit */
    const int*   a2f_dst  = (const int*)d_in[10];
    const int*   atom_b   = (const int*)d_in[11];
    const int*   fg_b     = (const int*)d_in[12];
    const float* eaW1 = (const float*)d_in[13]; const float* eab1 = (const float*)d_in[14];
    const float* eag1 = (const float*)d_in[15]; const float* eabe1= (const float*)d_in[16];
    const float* eaW2 = (const float*)d_in[17]; const float* eab2 = (const float*)d_in[18];
    const float* eag2 = (const float*)d_in[19]; const float* eabe2= (const float*)d_in[20];
    const float* efW1 = (const float*)d_in[21]; const float* efb1 = (const float*)d_in[22];
    const float* efg1 = (const float*)d_in[23]; const float* efbe1= (const float*)d_in[24];
    const float* efW2 = (const float*)d_in[25]; const float* efb2 = (const float*)d_in[26];
    const float* efg2 = (const float*)d_in[27]; const float* efbe2= (const float*)d_in[28];
    const float* edge_aW = (const float*)d_in[29]; const float* edge_ab = (const float*)d_in[30];
    const float* edge_fW = (const float*)d_in[31]; const float* edge_fb = (const float*)d_in[32];
    const float* gaWx = (const float*)d_in[33]; const float* gaWh = (const float*)d_in[34];
    const float* gabx = (const float*)d_in[35]; const float* gabh = (const float*)d_in[36];
    const float* gfWx = (const float*)d_in[37]; const float* gfWh = (const float*)d_in[38];
    const float* gfbx = (const float*)d_in[39]; const float* gfbh = (const float*)d_in[40];
    const float* scaW = (const float*)d_in[41]; const float* scab = (const float*)d_in[42];
    const float* scfW = (const float*)d_in[43]; const float* scfb = (const float*)d_in[44];
    const float* smW1 = (const float*)d_in[45]; const float* smb1 = (const float*)d_in[46];
    const float* smW2 = (const float*)d_in[47]; const float* smb2 = (const float*)d_in[48];
    const float* oaW1 = (const float*)d_in[49]; const float* oab1 = (const float*)d_in[50];
    const float* oaW2 = (const float*)d_in[51]; const float* oab2 = (const float*)d_in[52];
    const float* ofW1 = (const float*)d_in[53]; const float* ofb1 = (const float*)d_in[54];
    const float* ofW2 = (const float*)d_in[55]; const float* ofb2 = (const float*)d_in[56];
    float* out = (float*)d_out;

    // ---- workspace arena ----
    char* ws = (char*)d_ws;
    size_t off = 0;
    auto alloc = [&](size_t bytes) -> void* {
        void* p = ws + off;
        off = (off + bytes + 255) & ~(size_t)255;
        return p;
    };
    float* enc_tmp = (float*)alloc((size_t)NA * HID * 4);
    float* uaf     = (float*)alloc((size_t)NA * HID * 4);
    float* ufnf    = (float*)alloc((size_t)NF * HID * 4);
    float* uam     = (float*)alloc((size_t)NA * HID * 4);
    float* ufnm    = (float*)alloc((size_t)NF * HID * 4);
    float* aug     = (float*)alloc((size_t)NF * HID * 4);
    float* gx_a    = (float*)alloc((size_t)NA * 3 * HID * 4);
    float* gh_a    = (float*)alloc((size_t)NA * 3 * HID * 4);
    float* gx_f    = (float*)alloc((size_t)NF * 3 * HID * 4);
    float* gh_f    = (float*)alloc((size_t)NF * 3 * HID * 4);
    float* at_s    = (float*)alloc((size_t)NG * HID * 4);
    float* at_m    = (float*)alloc((size_t)NG * HID * 4);
    float* fg_s    = (float*)alloc((size_t)NG * HID * 4);
    float* fg_m    = (float*)alloc((size_t)NG * HID * 4);
    float* rep_a   = (float*)alloc((size_t)NG * REP * 4);
    float* rep_f   = (float*)alloc((size_t)NG * REP * 4);
    float* jbuf    = (float*)alloc((size_t)NG * 2 * 4);
    float* wbuf    = (float*)alloc((size_t)NG * 2 * 4);
    float* hid_a   = (float*)alloc((size_t)NG * 256 * 4);
    float* hid_f   = (float*)alloc((size_t)NG * 256 * 4);

    size_t f16_start = off;
    _Float16* af16   = (_Float16*)alloc((size_t)NA * 64 * 2);    // AIN=40 -> Kp 64
    _Float16* fnf16  = (_Float16*)alloc((size_t)NF * 96 * 2);    // FNIN=82 -> Kp 96
    _Float16* h16_a  = (_Float16*)alloc((size_t)NA * HID * 2);
    _Float16* h16_f  = (_Float16*)alloc((size_t)NF * HID * 2);
    _Float16* x16_a  = (_Float16*)alloc((size_t)NA * HID * 2);
    _Float16* cat16  = (_Float16*)alloc((size_t)NF * 2 * HID * 2);
    _Float16* wEA1   = (_Float16*)alloc((size_t)64 * HID * 2);           // packed
    _Float16* wEA2   = (_Float16*)alloc((size_t)HID * HID * 2);
    _Float16* wEF1   = (_Float16*)alloc((size_t)96 * HID * 2);
    _Float16* wEF2   = (_Float16*)alloc((size_t)HID * HID * 2);
    _Float16* wGAx   = (_Float16*)alloc((size_t)HID * 3 * HID * 2);
    _Float16* wGAh   = (_Float16*)alloc((size_t)HID * 3 * HID * 2);
    _Float16* wGFx   = (_Float16*)alloc((size_t)2 * HID * 3 * HID * 2);
    _Float16* wGFh   = (_Float16*)alloc((size_t)HID * 3 * HID * 2);
    _Float16* w2a    = (_Float16*)alloc((size_t)(BIN + 1) * HID * HID * 2);   // 1056x96
    _Float16* w2f    = (_Float16*)alloc((size_t)(FEIN + 1) * HID * HID * 2);  // 12576x96
    _Float16* wOA1   = (_Float16*)alloc((size_t)REPP * 256 * 2);
    _Float16* wOF1   = (_Float16*)alloc((size_t)REPP * 256 * 2);
    _Float16* wOA2   = (_Float16*)alloc((size_t)256 * 16 * 2);   // OUT=12 -> Np 16
    _Float16* wOF2   = (_Float16*)alloc((size_t)256 * 16 * 2);
    _Float16* rep16a = (_Float16*)alloc((size_t)NG * REPP * 2);
    _Float16* rep16f = (_Float16*)alloc((size_t)NG * REPP * 2);
    _Float16* hid16a = (_Float16*)alloc((size_t)NG * 256 * 2);
    _Float16* hid16f = (_Float16*)alloc((size_t)NG * 256 * 2);
    size_t f16_elems = (off - f16_start) / 2;

    // ---- host-side launch helpers ----
    auto fill32 = [&](float* p, float v, int n) {
        k_fill_f32<<<dim3((n + 255) / 256), dim3(256), 0, stream>>>(p, v, n);
    };
    auto cvt = [&](const float* s, _Float16* d, int M, int K, int lds, int ldd, int co) {
        int n = M * K;
        k_cvt_f16<<<dim3((n + 255) / 256), dim3(256), 0, stream>>>(s, d, M, K, lds, ldd, co);
    };
    auto pack = [&](const float* s, _Float16* d, int K, int N, int Kp, int Np) {
        int n = Kp * Np;
        k_pack_b<<<dim3((n + 255) / 256), dim3(256), 0, stream>>>(s, d, K, N, Kp, Np);
    };
    auto gemm = [&](const _Float16* A, int lda, const _Float16* Bp,
                    const float* bias, const float* gamma, const float* beta,
                    float* C, int ldc, int M, int Npad, int Nreal, int K, int relu) {
        dim3 g(Npad / 16, M / 64), b(128);
        k_wmma_gemm<<<g, b, 0, stream>>>(A, lda, Bp, bias, gamma, beta, C, ldc, K, Nreal, relu);
    };

    // ---- zero f16 arena (A-side K-padding regions must be 0) ----
    k_fill_f16<<<dim3((int)((f16_elems + 255) / 256)), dim3(256), 0, stream>>>(
        (_Float16*)(ws + f16_start), (int)f16_elems);

    // ---- weight staging: f32 -> packed fragment-major f16 ----
    pack(eaW1, wEA1, AIN, HID, 64, HID);
    pack(eaW2, wEA2, HID, HID, HID, HID);
    pack(efW1, wEF1, FNIN, HID, 96, HID);
    pack(efW2, wEF2, HID, HID, HID, HID);
    pack(gaWx, wGAx, HID, 3 * HID, HID, 3 * HID);
    pack(gaWh, wGAh, HID, 3 * HID, HID, 3 * HID);
    pack(gfWx, wGFx, 2 * HID, 3 * HID, 2 * HID, 3 * HID);
    pack(gfWh, wGFh, HID, 3 * HID, HID, 3 * HID);
    pack(oaW1, wOA1, REP, 256, REPP, 256);
    pack(ofW1, wOF1, REP, 256, REPP, 256);
    pack(oaW2, wOA2, 256, OUTD, 256, 16);
    pack(ofW2, wOF2, 256, OUTD, 256, 16);
    {
        int na = (BIN + 1) * HID * HID, nf = (FEIN + 1) * HID * HID;
        k_pack_edgeW<<<dim3((na + 255) / 256), dim3(256), 0, stream>>>(edge_aW, edge_ab, w2a, BIN);
        k_pack_edgeW<<<dim3((nf + 255) / 256), dim3(256), 0, stream>>>(edge_fW, edge_fb, w2f, FEIN);
    }
    cvt(af, af16, NA, AIN, AIN, 64, 0);
    cvt(fnf, fnf16, NF, FNIN, FNIN, 96, 0);

    // ---- encoders: relu((x@W1+b1)*g1+be1) -> relu((h@W2+b2)*g2+be2) ----
    gemm(af16, 64, wEA1, eab1, eag1, eabe1, enc_tmp, HID, NA, HID, HID, 64, 1);
    cvt(enc_tmp, h16_a, NA, HID, HID, HID, 0);
    gemm(h16_a, HID, wEA2, eab2, eag2, eabe2, uaf, HID, NA, HID, HID, HID, 1);

    gemm(fnf16, 96, wEF1, efb1, efg1, efbe1, enc_tmp, HID, NF, HID, HID, 96, 1);
    cvt(enc_tmp, h16_f, NF, HID, HID, HID, 0);
    gemm(h16_f, HID, wEF2, efb2, efg2, efbe2, ufnf, HID, NF, HID, HID, HID, 1);

    // ---- message passing steps ----
    for (int step = 0; step < 2; ++step) {
        cvt(uaf, h16_a, NA, HID, HID, HID, 0);
        cvt(ufnf, h16_f, NF, HID, HID, HID, 0);

        fill32(ufnm, 0.f, NF * HID);
        fill32(uam, 0.f, NA * HID);
        fill32(aug, 0.f, NF * HID);

        // fused NNConv messages (WMMA, A generated in registers)
        k_nnconv_msg<false><<<dim3(HID / 16, NEF / 64), dim3(128), 0, stream>>>(
            fef, FEIN, h16_f, f_src, f_dst, w2f, ufnm);
        k_nnconv_msg<true><<<dim3(HID / 16, NB / 64), dim3(128), 0, stream>>>(
            bf, BIN, h16_a, a_src, a_dst, w2a, uam);

        // atom -> motif aggregation (a2f_src == arange(NA))
        k_scatter_add_rows<<<dim3((NA * HID + 255) / 256), dim3(256), 0, stream>>>(
            uam, a2f_dst, aug, NA);

        // fg GRU: x = [ufnm || aug]
        cvt(ufnm, cat16, NF, HID, HID, 2 * HID, 0);
        cvt(aug, cat16, NF, HID, HID, 2 * HID, HID);
        gemm(cat16, 2 * HID, wGFx, gfbx, nullptr, nullptr,
             gx_f, 3 * HID, NF, 3 * HID, 3 * HID, 2 * HID, 0);
        gemm(h16_f, HID, wGFh, gfbh, nullptr, nullptr,
             gh_f, 3 * HID, NF, 3 * HID, 3 * HID, HID, 0);
        k_gru<<<dim3((NF * HID + 255) / 256), dim3(256), 0, stream>>>(gx_f, gh_f, ufnf, NF);

        // atom GRU: x = uam
        cvt(uam, x16_a, NA, HID, HID, HID, 0);
        gemm(x16_a, HID, wGAx, gabx, nullptr, nullptr,
             gx_a, 3 * HID, NA, 3 * HID, 3 * HID, HID, 0);
        gemm(h16_a, HID, wGAh, gabh, nullptr, nullptr,
             gh_a, 3 * HID, NA, 3 * HID, 3 * HID, HID, 0);
        k_gru<<<dim3((NA * HID + 255) / 256), dim3(256), 0, stream>>>(gx_a, gh_a, uaf, NA);
    }

    // ---- readouts ----
    fill32(at_s, 0.f, NG * HID);
    fill32(fg_s, 0.f, NG * HID);
    fill32(at_m, -__builtin_huge_valf(), NG * HID);
    fill32(fg_m, -__builtin_huge_valf(), NG * HID);
    k_segment_reduce<<<dim3((NA * HID + 255) / 256), dim3(256), 0, stream>>>(uaf, atom_b, NA, at_s, at_m);
    k_segment_reduce<<<dim3((NF * HID + 255) / 256), dim3(256), 0, stream>>>(ufnf, fg_b, NF, fg_s, fg_m);
    k_build_rep<<<dim3((NG * REP + 255) / 256), dim3(256), 0, stream>>>(at_s, at_m, mf, rep_a);
    k_build_rep<<<dim3((NG * REP + 255) / 256), dim3(256), 0, stream>>>(fg_s, fg_m, mf, rep_f);

    // ---- contextual rescale ----
    k_scale_dot<<<dim3(1), dim3(NG), 0, stream>>>(rep_a, scaW, scab, jbuf, 0);
    k_scale_dot<<<dim3(1), dim3(NG), 0, stream>>>(rep_f, scfW, scfb, jbuf, 1);
    k_scale_mlp<<<dim3(1), dim3(NG), 0, stream>>>(jbuf, smW1, smb1, smW2, smb2, wbuf);
    k_scale_apply<<<dim3((NG * REP + 255) / 256), dim3(256), 0, stream>>>(rep_a, wbuf, 0);
    k_scale_apply<<<dim3((NG * REP + 255) / 256), dim3(256), 0, stream>>>(rep_f, wbuf, 1);

    // ---- output heads ----
    cvt(rep_a, rep16a, NG, REP, REP, REPP, 0);
    cvt(rep_f, rep16f, NG, REP, REP, REPP, 0);
    gemm(rep16a, REPP, wOA1, oab1, nullptr, nullptr, hid_a, 256, NG, 256, 256, REPP, 1);
    gemm(rep16f, REPP, wOF1, ofb1, nullptr, nullptr, hid_f, 256, NG, 256, 256, REPP, 1);
    cvt(hid_a, hid16a, NG, 256, 256, 256, 0);
    cvt(hid_f, hid16f, NG, 256, 256, 256, 0);
    gemm(hid16a, 256, wOA2, oab2, nullptr, nullptr, out, OUTD, NG, 16, OUTD, 256, 0);
    gemm(hid16f, 256, wOF2, ofb2, nullptr, nullptr, out + NG * OUTD, OUTD, NG, 16, OUTD, 256, 0);
}
